// TransformerLayer_42923903156585
// MI455X (gfx1250) — matrix operand order repaired and checked by reading, not compiled, not run on hardware
//
#include <hip/hip_runtime.h>
#include <hip/hip_bf16.h>
#include <math.h>

// ---------------------------------------------------------------------------
// Problem constants (from reference): B=8, C=512, T=1024, H=8, CK=64, HCH=2048
// ---------------------------------------------------------------------------
#define Bn   8
#define Cn   512
#define Tn   1024
#define Hn   8
#define CKn  64
#define HCHn 2048

typedef __bf16 bf16;
typedef __attribute__((ext_vector_type(16))) __bf16 v16bf;
typedef __attribute__((ext_vector_type(8)))  float  v8f;

#define LDK32 40   // LDS row stride (elems) for K=32 tiles  (bank-conflict free)
#define LDK64 72   // LDS row stride (elems) for K=64 tiles
#define LDF   67   // LDS row stride (f32) for epilogue transpose buffer

// ---------------------------------------------------------------------------
// WMMA helpers (CDNA5 gfx1250, wave32). Fragment layouts per ISA 7.12.2:
//  A (16x32 bf16): lane lo=l&15 -> row M=lo.  hi=l>>4 selects K-interleave:
//    vgpr 0..3 : K = (hi?8:0)  + 2*vg + p
//    vgpr 4..7 : K = 16+(hi?8:0)+2*(vg-4)+p
//  B (32x16 bf16): lane lo -> col N=lo. same K-interleave as A
//  C/D (16x16 f32): vgpr r: row M = r + 8*hi, col N = lo
// ---------------------------------------------------------------------------
__device__ __forceinline__ v16bf load_fragA(const bf16* A, int ld, int m0, int k0) {
  const int lane = threadIdx.x & 31;
  const int lo = lane & 15, hi = lane >> 4;
  v16bf a;
#pragma unroll
  for (int e = 0; e < 16; ++e) {
    const int vg = e >> 1;
    const int k = ((vg & 4) << 2) + (hi << 3) + ((vg & 3) << 1) + (e & 1);
    a[e] = A[(m0 + lo) * ld + k0 + k];
  }
  return a;
}

__device__ __forceinline__ v16bf load_fragB(const bf16* Bm, int ld, int n0, int k0) {
  const int lane = threadIdx.x & 31;
  const int lo = lane & 15, hi = lane >> 4;
  v16bf b;
#pragma unroll
  for (int e = 0; e < 16; ++e) {
    const int vg = e >> 1;
    const int k = ((vg & 4) << 2) + (hi << 3) + ((vg & 3) << 1) + (e & 1);
    b[e] = Bm[(n0 + lo) * ld + k0 + k];
  }
  return b;
}

__device__ __forceinline__ v8f wmma_bf16(v16bf a, v16bf b, v8f c) {
  return __builtin_amdgcn_wmma_f32_16x16x32_bf16(false, a, false, b,
                                                 (short)0, c, false, false);
}

__device__ __forceinline__ void store_acc(float* Sb, int m0, int n0, v8f a) {
  const int lane = threadIdx.x & 31;
  const int lo = lane & 15, hi = lane >> 4;
#pragma unroll
  for (int r = 0; r < 8; ++r)
    Sb[(m0 + r + (hi << 3)) * LDF + n0 + lo] = a[r];
}

// ---------------------------------------------------------------------------
// Prep kernels: f32 -> bf16 conversion, conv-weight repack (O,Cin,3)->(3,O,Cin)
// ---------------------------------------------------------------------------
__global__ void k_f2bf(const float* __restrict__ in, bf16* __restrict__ out, int n) {
  int i = blockIdx.x * blockDim.x + threadIdx.x;
  if (i < n) out[i] = (bf16)in[i];
}

__global__ void k_pack3(const float* __restrict__ in, bf16* __restrict__ out,
                        int M, int Cin) {
  int i = blockIdx.x * blockDim.x + threadIdx.x;
  int tot = 3 * M * Cin;
  if (i >= tot) return;
  int dk = i / (M * Cin);
  int r  = i - dk * M * Cin;
  int o  = r / Cin;
  int c  = r - o * Cin;
  out[i] = (bf16)in[((size_t)o * Cin + c) * 3 + dk];
}

// ---------------------------------------------------------------------------
// QKV: per (b,h) GEMM w[h](64x64) @ x[h](64xT), bias + RoPE epilogue.
//  q,k -> (B,H,T,CK) bf16 (t-major, for A/B frags of attention)
//  v   -> (B,H,CK,T) bf16 (c-major, B frag of PV gemm)
// ---------------------------------------------------------------------------
__global__ __launch_bounds__(256) void k_qkv(
    const bf16* __restrict__ xb, const bf16* __restrict__ wqb,
    const bf16* __restrict__ wkb, const bf16* __restrict__ wvb,
    const float* __restrict__ bq, const float* __restrict__ bk,
    const float* __restrict__ bv,
    bf16* __restrict__ qt, bf16* __restrict__ kt, bf16* __restrict__ vt)
{
  __shared__ bf16 Aq[64 * LDK64];
  __shared__ bf16 Ak[64 * LDK64];
  __shared__ bf16 Av[64 * LDK64];
  __shared__ bf16 Bx[64 * LDK64];
  __shared__ float Fb[64 * LDF];

  const int blk = blockIdx.x;
  const int t0 = (blk & 15) << 6;
  const int h  = (blk >> 4) & 7;
  const int b  = blk >> 7;
  const int tid = threadIdx.x;

  for (int i = 0; i < 16; ++i) {
    int idx = tid + (i << 8);
    int m = idx >> 6, k = idx & 63;
    int g = (h * 64 + m) * 64 + k;
    Aq[m * LDK64 + k] = wqb[g];
    Ak[m * LDK64 + k] = wkb[g];
    Av[m * LDK64 + k] = wvb[g];
    // x tile, transposed into [t][c] so B-frag K-pairs are contiguous
    Bx[k * LDK64 + m] = xb[((size_t)b * Cn + h * 64 + m) * Tn + t0 + k];
  }
  __syncthreads();

  const int lane = tid & 31;
  const int lo = lane & 15, hi = lane >> 4;
  const int wave = tid >> 5;
  const int wm = (wave >> 1) << 4;   // 0,16,32,48
  const int wn = (wave & 1) << 5;    // 0,32

  v8f qa0 = {}, qa1 = {}, ka0 = {}, ka1 = {}, va0 = {}, va1 = {};
#pragma unroll
  for (int ks = 0; ks < 64; ks += 32) {
    v16bf aq = load_fragA(Aq, LDK64, wm, ks);
    v16bf ak = load_fragA(Ak, LDK64, wm, ks);
    v16bf av = load_fragA(Av, LDK64, wm, ks);
    v16bf b0 = load_fragB(Bx, LDK64, wn, ks);
    v16bf b1 = load_fragB(Bx, LDK64, wn + 16, ks);
    qa0 = wmma_bf16(aq, b0, qa0); qa1 = wmma_bf16(aq, b1, qa1);
    ka0 = wmma_bf16(ak, b0, ka0); ka1 = wmma_bf16(ak, b1, ka1);
    va0 = wmma_bf16(av, b0, va0); va1 = wmma_bf16(av, b1, va1);
  }

  // ---- Q: bias + RoPE -> qt (B,H,T,CK) ----
#pragma unroll
  for (int r = 0; r < 8; ++r) {
    float bias = bq[h * 64 + wm + r + (hi << 3)];
    qa0[r] += bias; qa1[r] += bias;
  }
  store_acc(Fb, wm, wn, qa0);
  store_acc(Fb, wm, wn + 16, qa1);
  __syncthreads();
  for (int i = 0; i < 16; ++i) {
    int idx = tid + (i << 8);
    int t = idx >> 6, j = idx & 63;
    float val;
    if (j < 32) {
      int f = j & 15;
      float ang = (float)(t0 + t) * __expf(-(float)f * 0.5756462732485114f);
      float sn, cs; __sincosf(ang, &sn, &cs);
      float x1 = Fb[f * LDF + t], x2 = Fb[(f + 16) * LDF + t];
      val = (j < 16) ? (x1 * cs - x2 * sn) : (x1 * sn + x2 * cs);
    } else {
      val = Fb[j * LDF + t];
    }
    qt[(((size_t)b * Hn + h) * Tn + t0 + t) * 64 + j] = (bf16)val;
  }
  __syncthreads();

  // ---- K: bias + RoPE -> kt (B,H,T,CK) ----
#pragma unroll
  for (int r = 0; r < 8; ++r) {
    float bias = bk[h * 64 + wm + r + (hi << 3)];
    ka0[r] += bias; ka1[r] += bias;
  }
  store_acc(Fb, wm, wn, ka0);
  store_acc(Fb, wm, wn + 16, ka1);
  __syncthreads();
  for (int i = 0; i < 16; ++i) {
    int idx = tid + (i << 8);
    int t = idx >> 6, j = idx & 63;
    float val;
    if (j < 32) {
      int f = j & 15;
      float ang = (float)(t0 + t) * __expf(-(float)f * 0.5756462732485114f);
      float sn, cs; __sincosf(ang, &sn, &cs);
      float x1 = Fb[f * LDF + t], x2 = Fb[(f + 16) * LDF + t];
      val = (j < 16) ? (x1 * cs - x2 * sn) : (x1 * sn + x2 * cs);
    } else {
      val = Fb[j * LDF + t];
    }
    kt[(((size_t)b * Hn + h) * Tn + t0 + t) * 64 + j] = (bf16)val;
  }
  __syncthreads();

  // ---- V: bias -> vt (B,H,CK,T) ----
#pragma unroll
  for (int r = 0; r < 8; ++r) {
    float bias = bv[h * 64 + wm + r + (hi << 3)];
    va0[r] += bias; va1[r] += bias;
  }
  store_acc(Fb, wm, wn, va0);
  store_acc(Fb, wm, wn + 16, va1);
  __syncthreads();
  for (int i = 0; i < 16; ++i) {
    int idx = tid + (i << 8);
    int j = idx >> 6, t = idx & 63;
    vt[(((size_t)b * Hn + h) * 64 + j) * Tn + t0 + t] = (bf16)Fb[j * LDF + t];
  }
}

// ---------------------------------------------------------------------------
// Flash attention: per (b,h, 64 q-rows). S=QK^T/8, mask, online softmax,
// O += P V. Output att (B,T,C) bf16 with c_global = h*64 + ck.
// ---------------------------------------------------------------------------
__global__ __launch_bounds__(256) void k_attn(
    const bf16* __restrict__ qt, const bf16* __restrict__ kt,
    const bf16* __restrict__ vt, const int* __restrict__ amask,
    bf16* __restrict__ att)
{
  __shared__ bf16 Qs[64 * LDK64];
  __shared__ bf16 Bs[64 * LDK64];   // K tile, then V tile
  __shared__ bf16 Ps[64 * LDK64];   // P (A-frag layout [t][s])
  __shared__ float Sb[64 * LDF];
  __shared__ float mrow[64], lrow[64], crow[64];

  const int blk = blockIdx.x;
  const int t0 = (blk & 15) << 6;
  const int h  = (blk >> 4) & 7;
  const int b  = blk >> 7;
  const int tid = threadIdx.x;

  const bf16* qbase = qt + (((size_t)b * Hn + h) * Tn) * 64;
  const bf16* kbase = kt + (((size_t)b * Hn + h) * Tn) * 64;
  const bf16* vbase = vt + (((size_t)b * Hn + h) * 64) * Tn;

  for (int i = 0; i < 16; ++i) {
    int idx = tid + (i << 8);
    int t = idx >> 6, c = idx & 63;
    Qs[t * LDK64 + c] = qbase[(size_t)(t0 + t) * 64 + c];
  }
  if (tid < 64) { mrow[tid] = -3.0e38f; lrow[tid] = 0.0f; }

  const int lane = tid & 31;
  const int lo = lane & 15, hi = lane >> 4;
  const int wave = tid >> 5;
  const int wm = (wave >> 1) << 4;
  const int wn = (wave & 1) << 5;

  v8f o0 = {}, o1 = {};
  __syncthreads();

  for (int s0 = 0; s0 < Tn; s0 += 64) {
    // stage K tile [s][c]
    for (int i = 0; i < 16; ++i) {
      int idx = tid + (i << 8);
      int s = idx >> 6, c = idx & 63;
      Bs[s * LDK64 + c] = kbase[(size_t)(s0 + s) * 64 + c];
    }
    __syncthreads();

    v8f sc0 = {}, sc1 = {};
#pragma unroll
    for (int ks = 0; ks < 64; ks += 32) {
      v16bf a  = load_fragA(Qs, LDK64, wm, ks);
      v16bf b0 = load_fragB(Bs, LDK64, wn, ks);
      v16bf b1 = load_fragB(Bs, LDK64, wn + 16, ks);
      sc0 = wmma_bf16(a, b0, sc0);
      sc1 = wmma_bf16(a, b1, sc1);
    }
#pragma unroll
    for (int r = 0; r < 8; ++r) { sc0[r] *= 0.125f; sc1[r] *= 0.125f; }
    store_acc(Sb, wm, wn, sc0);
    store_acc(Sb, wm, wn + 16, sc1);
    __syncthreads();

    // online softmax, one thread per q-row
    if (tid < 64) {
      const int t = tid;
      const int* mp = amask + ((size_t)b * Tn + (t0 + t)) * Tn + s0;
      float mold = mrow[t];
      float mx = mold;
      for (int s = 0; s < 64; ++s) {
        float xv = Sb[t * LDF + s];
        if (mp[s] == 0) xv = -10000.0f;
        Sb[t * LDF + s] = xv;
        mx = fmaxf(mx, xv);
      }
      float corr = __expf(mold - mx);
      float l = lrow[t] * corr;
      for (int s = 0; s < 64; ++s) {
        float p = __expf(Sb[t * LDF + s] - mx);
        l += p;
        Ps[t * LDK64 + s] = (bf16)p;
      }
      mrow[t] = mx; lrow[t] = l; crow[t] = corr;
    }
    __syncthreads();

    // rescale O accumulators
#pragma unroll
    for (int r = 0; r < 8; ++r) {
      float cr = crow[wm + r + (hi << 3)];
      o0[r] *= cr; o1[r] *= cr;
    }
    // stage V tile [c][s]
    for (int i = 0; i < 16; ++i) {
      int idx = tid + (i << 8);
      int c = idx >> 6, s = idx & 63;
      Bs[c * LDK64 + s] = vbase[(size_t)c * Tn + s0 + s];
    }
    __syncthreads();

#pragma unroll
    for (int ks = 0; ks < 64; ks += 32) {
      v16bf a  = load_fragA(Ps, LDK64, wm, ks);
      v16bf b0 = load_fragB(Bs, LDK64, wn, ks);
      v16bf b1 = load_fragB(Bs, LDK64, wn + 16, ks);
      o0 = wmma_bf16(a, b0, o0);
      o1 = wmma_bf16(a, b1, o1);
    }
    __syncthreads();
  }

#pragma unroll
  for (int r = 0; r < 8; ++r) {
    float inv = 1.0f / lrow[wm + r + (hi << 3)];
    o0[r] *= inv; o1[r] *= inv;
  }
  store_acc(Sb, wm, wn, o0);
  store_acc(Sb, wm, wn + 16, o1);
  __syncthreads();
  for (int i = 0; i < 16; ++i) {
    int idx = tid + (i << 8);
    int t = idx >> 6, c = idx & 63;
    att[((size_t)b * Tn + t0 + t) * Cn + h * 64 + c] = (bf16)Sb[t * LDF + c];
  }
}

// ---------------------------------------------------------------------------
// W_o GEMM: s1[b,t,o] = wo @ att + bo + x (residual). K=512.
// ---------------------------------------------------------------------------
__global__ __launch_bounds__(256) void k_wogemm(
    const bf16* __restrict__ wob, const bf16* __restrict__ att,
    const float* __restrict__ bo, const float* __restrict__ x,
    float* __restrict__ s1)
{
  __shared__ bf16 As[64 * LDK32];
  __shared__ bf16 Bsh[64 * LDK32];
  __shared__ float Fb[64 * LDF];

  const int blk = blockIdx.x;
  const int t0 = (blk & 15) << 6;
  const int mo = ((blk >> 4) & 7) << 6;
  const int b  = blk >> 7;
  const int tid = threadIdx.x;
  const int lane = tid & 31;
  const int lo = lane & 15, hi = lane >> 4;
  const int wave = tid >> 5;
  const int wm = (wave >> 1) << 4;
  const int wn = (wave & 1) << 5;

  v8f a0 = {}, a1 = {};
  for (int k0 = 0; k0 < Cn; k0 += 32) {
    for (int i = 0; i < 8; ++i) {
      int idx = tid + (i << 8);
      int m = idx >> 5, k = idx & 31;
      As[m * LDK32 + k]  = wob[(size_t)(mo + m) * Cn + k0 + k];
      Bsh[m * LDK32 + k] = att[((size_t)b * Tn + t0 + m) * Cn + k0 + k];
    }
    __syncthreads();
    v16bf af = load_fragA(As, LDK32, wm, 0);
    v16bf b0 = load_fragB(Bsh, LDK32, wn, 0);
    v16bf b1 = load_fragB(Bsh, LDK32, wn + 16, 0);
    a0 = wmma_bf16(af, b0, a0);
    a1 = wmma_bf16(af, b1, a1);
    __syncthreads();
  }
#pragma unroll
  for (int r = 0; r < 8; ++r) {
    int m = wm + r + (hi << 3);
    float bias = bo[mo + m];
    a0[r] += bias + x[((size_t)b * Cn + mo + m) * Tn + t0 + wn + lo];
    a1[r] += bias + x[((size_t)b * Cn + mo + m) * Tn + t0 + wn + 16 + lo];
  }
  store_acc(Fb, wm, wn, a0);
  store_acc(Fb, wm, wn + 16, a1);
  __syncthreads();
  for (int i = 0; i < 16; ++i) {
    int idx = tid + (i << 8);
    int t = idx >> 6, o = idx & 63;
    s1[((size_t)b * Tn + t0 + t) * Cn + mo + o] = Fb[o * LDF + t];
  }
}

// ---------------------------------------------------------------------------
// LayerNorm over C per (b,t). LN1 emits h (f32 residual copy) + h*mask (bf16).
// ---------------------------------------------------------------------------
__global__ __launch_bounds__(256) void k_ln1(
    const float* __restrict__ s1, const float* __restrict__ g,
    const float* __restrict__ beta, const float* __restrict__ xmask,
    bf16* __restrict__ hb, float* __restrict__ h32)
{
  __shared__ float red1[8], red2[8];
  const int bt = blockIdx.x;
  const int tid = threadIdx.x;
  const float* row = s1 + (size_t)bt * Cn;
  float v0 = row[tid], v1 = row[tid + 256];
  float s = v0 + v1, ss = v0 * v0 + v1 * v1;
  for (int off = 16; off; off >>= 1) {
    s  += __shfl_xor(s, off);
    ss += __shfl_xor(ss, off);
  }
  if ((tid & 31) == 0) { red1[tid >> 5] = s; red2[tid >> 5] = ss; }
  __syncthreads();
  if (tid == 0) {
    float a = 0.f, c = 0.f;
    for (int w = 0; w < 8; ++w) { a += red1[w]; c += red2[w]; }
    red1[0] = a; red2[0] = c;
  }
  __syncthreads();
  float mean = red1[0] * (1.0f / Cn);
  float var  = red2[0] * (1.0f / Cn) - mean * mean;
  float rs = rsqrtf(var + 1e-5f);
  float mk = xmask[bt];
  float y0 = (v0 - mean) * rs * g[tid] + beta[tid];
  float y1 = (v1 - mean) * rs * g[tid + 256] + beta[tid + 256];
  h32[(size_t)bt * Cn + tid]       = y0;
  h32[(size_t)bt * Cn + tid + 256] = y1;
  hb[(size_t)bt * Cn + tid]        = (bf16)(y0 * mk);
  hb[(size_t)bt * Cn + tid + 256]  = (bf16)(y1 * mk);
}

// ---------------------------------------------------------------------------
// conv1: f[b,t,o] = gelu( sum_dk W1p[dk] @ h[:, t+dk-1] + b1 ) * mask
// M=2048, K=3*512.  A from packed (3,HCH,C) bf16 weights.
// ---------------------------------------------------------------------------
__global__ __launch_bounds__(256) void k_conv1(
    const bf16* __restrict__ w1p, const bf16* __restrict__ hb,
    const float* __restrict__ b1, const float* __restrict__ xmask,
    bf16* __restrict__ fbuf)
{
  __shared__ bf16 As[64 * LDK32];
  __shared__ bf16 Bsh[64 * LDK32];
  __shared__ float Fb[64 * LDF];

  const int blk = blockIdx.x;
  const int t0 = (blk & 15) << 6;
  const int mo = ((blk >> 4) & 31) << 6;
  const int b  = blk >> 9;
  const int tid = threadIdx.x;
  const int lane = tid & 31;
  const int lo = lane & 15, hi = lane >> 4;
  const int wave = tid >> 5;
  const int wm = (wave >> 1) << 4;
  const int wn = (wave & 1) << 5;

  v8f a0 = {}, a1 = {};
  for (int dk = 0; dk < 3; ++dk) {
    const bf16* Aw = w1p + (size_t)dk * HCHn * Cn + (size_t)mo * Cn;
    for (int k0 = 0; k0 < Cn; k0 += 32) {
      for (int i = 0; i < 8; ++i) {
        int idx = tid + (i << 8);
        int m = idx >> 5, k = idx & 31;
        As[m * LDK32 + k] = Aw[(size_t)m * Cn + k0 + k];
        int tg = t0 + m + dk - 1;
        Bsh[m * LDK32 + k] = (tg >= 0 && tg < Tn)
            ? hb[((size_t)b * Tn + tg) * Cn + k0 + k] : (bf16)0.0f;
      }
      __syncthreads();
      v16bf af = load_fragA(As, LDK32, wm, 0);
      v16bf b0 = load_fragB(Bsh, LDK32, wn, 0);
      v16bf b1f = load_fragB(Bsh, LDK32, wn + 16, 0);
      a0 = wmma_bf16(af, b0, a0);
      a1 = wmma_bf16(af, b1f, a1);
      __syncthreads();
    }
  }
#pragma unroll
  for (int r = 0; r < 8; ++r) {
    int m = wm + r + (hi << 3);
    float bias = b1[mo + m];
    float g0 = a0[r] + bias;
    float g1 = a1[r] + bias;
    g0 = 0.5f * g0 * (1.0f + erff(g0 * 0.70710678118f));
    g1 = 0.5f * g1 * (1.0f + erff(g1 * 0.70710678118f));
    g0 *= xmask[(size_t)b * Tn + t0 + wn + lo];
    g1 *= xmask[(size_t)b * Tn + t0 + wn + 16 + lo];
    a0[r] = g0; a1[r] = g1;
  }
  store_acc(Fb, wm, wn, a0);
  store_acc(Fb, wm, wn + 16, a1);
  __syncthreads();
  for (int i = 0; i < 16; ++i) {
    int idx = tid + (i << 8);
    int t = idx >> 6, o = idx & 63;
    fbuf[((size_t)b * Tn + t0 + t) * HCHn + mo + o] = (bf16)Fb[o * LDF + t];
  }
}

// ---------------------------------------------------------------------------
// conv2: s2[b,t,o] = ( sum_dk W2p[dk] @ f[:, t+dk-1] + b2 ) * mask + h
// M=512, K=3*2048.
// ---------------------------------------------------------------------------
__global__ __launch_bounds__(256) void k_conv2(
    const bf16* __restrict__ w2p, const bf16* __restrict__ fbuf,
    const float* __restrict__ b2, const float* __restrict__ xmask,
    const float* __restrict__ h32, float* __restrict__ s2)
{
  __shared__ bf16 As[64 * LDK32];
  __shared__ bf16 Bsh[64 * LDK32];
  __shared__ float Fb[64 * LDF];

  const int blk = blockIdx.x;
  const int t0 = (blk & 15) << 6;
  const int mo = ((blk >> 4) & 7) << 6;
  const int b  = blk >> 7;
  const int tid = threadIdx.x;
  const int lane = tid & 31;
  const int hi = lane >> 4;
  const int wave = tid >> 5;
  const int wm = (wave >> 1) << 4;
  const int wn = (wave & 1) << 5;

  v8f a0 = {}, a1 = {};
  for (int dk = 0; dk < 3; ++dk) {
    const bf16* Aw = w2p + (size_t)dk * Cn * HCHn + (size_t)mo * HCHn;
    for (int k0 = 0; k0 < HCHn; k0 += 32) {
      for (int i = 0; i < 8; ++i) {
        int idx = tid + (i << 8);
        int m = idx >> 5, k = idx & 31;
        As[m * LDK32 + k] = Aw[(size_t)m * HCHn + k0 + k];
        int tg = t0 + m + dk - 1;
        Bsh[m * LDK32 + k] = (tg >= 0 && tg < Tn)
            ? fbuf[((size_t)b * Tn + tg) * HCHn + k0 + k] : (bf16)0.0f;
      }
      __syncthreads();
      v16bf af = load_fragA(As, LDK32, wm, 0);
      v16bf b0 = load_fragB(Bsh, LDK32, wn, 0);
      v16bf b1f = load_fragB(Bsh, LDK32, wn + 16, 0);
      a0 = wmma_bf16(af, b0, a0);
      a1 = wmma_bf16(af, b1f, a1);
      __syncthreads();
    }
  }
#pragma unroll
  for (int r = 0; r < 8; ++r) {
    float bias = b2[mo + wm + r + (hi << 3)];
    a0[r] += bias; a1[r] += bias;
  }
  store_acc(Fb, wm, wn, a0);
  store_acc(Fb, wm, wn + 16, a1);
  __syncthreads();
  for (int i = 0; i < 16; ++i) {
    int idx = tid + (i << 8);
    int t = idx >> 6, o = idx & 63;
    float mk = xmask[(size_t)b * Tn + t0 + t];
    float v = Fb[o * LDF + t] * mk
            + h32[((size_t)b * Tn + t0 + t) * Cn + mo + o];
    s2[((size_t)b * Tn + t0 + t) * Cn + mo + o] = v;
  }
}

// ---------------------------------------------------------------------------
// LN2 -> final output (B,C,T) f32
// ---------------------------------------------------------------------------
__global__ __launch_bounds__(256) void k_ln2(
    const float* __restrict__ s2, const float* __restrict__ g,
    const float* __restrict__ beta, float* __restrict__ out)
{
  __shared__ float red1[8], red2[8];
  const int bt = blockIdx.x;
  const int tid = threadIdx.x;
  const float* row = s2 + (size_t)bt * Cn;
  float v0 = row[tid], v1 = row[tid + 256];
  float s = v0 + v1, ss = v0 * v0 + v1 * v1;
  for (int off = 16; off; off >>= 1) {
    s  += __shfl_xor(s, off);
    ss += __shfl_xor(ss, off);
  }
  if ((tid & 31) == 0) { red1[tid >> 5] = s; red2[tid >> 5] = ss; }
  __syncthreads();
  if (tid == 0) {
    float a = 0.f, c = 0.f;
    for (int w = 0; w < 8; ++w) { a += red1[w]; c += red2[w]; }
    red1[0] = a; red2[0] = c;
  }
  __syncthreads();
  float mean = red1[0] * (1.0f / Cn);
  float var  = red2[0] * (1.0f / Cn) - mean * mean;
  float rs = rsqrtf(var + 1e-5f);
  const int b = bt >> 10, t = bt & (Tn - 1);
  float y0 = (v0 - mean) * rs * g[tid] + beta[tid];
  float y1 = (v1 - mean) * rs * g[tid + 256] + beta[tid + 256];
  out[((size_t)b * Cn + tid) * Tn + t]         = y0;
  out[((size_t)b * Cn + tid + 256) * Tn + t]   = y1;
}

// ---------------------------------------------------------------------------
// Host launcher
// ---------------------------------------------------------------------------
extern "C" void kernel_launch(void* const* d_in, const int* in_sizes, int n_in,
                              void* d_out, int out_size, void* d_ws, size_t ws_size,
                              hipStream_t stream) {
  (void)in_sizes; (void)n_in; (void)out_size; (void)ws_size;
  const float* x     = (const float*)d_in[0];
  const float* xmask = (const float*)d_in[1];
  const int*   amask = (const int*)d_in[2];
  const float* wq = (const float*)d_in[3];
  const float* bq = (const float*)d_in[4];
  const float* wk = (const float*)d_in[5];
  const float* bk = (const float*)d_in[6];
  const float* wv = (const float*)d_in[7];
  const float* bv = (const float*)d_in[8];
  const float* wo = (const float*)d_in[9];
  const float* bo = (const float*)d_in[10];
  const float* w1 = (const float*)d_in[11];
  const float* b1 = (const float*)d_in[12];
  const float* w2 = (const float*)d_in[13];
  const float* b2 = (const float*)d_in[14];
  const float* g1 = (const float*)d_in[15];
  const float* be1 = (const float*)d_in[16];
  const float* g2 = (const float*)d_in[17];
  const float* be2 = (const float*)d_in[18];
  float* out = (float*)d_out;

  char* ws = (char*)d_ws;
  size_t off = 0;
  auto alloc = [&](size_t bytes) { size_t r = off; off += (bytes + 255) & ~(size_t)255; return r; };

  bf16* xb   = (bf16*)(ws + alloc((size_t)Bn * Cn * Tn * 2));
  bf16* wqb  = (bf16*)(ws + alloc((size_t)Cn * CKn * 2));
  bf16* wkb  = (bf16*)(ws + alloc((size_t)Cn * CKn * 2));
  bf16* wvb  = (bf16*)(ws + alloc((size_t)Cn * CKn * 2));
  bf16* wob  = (bf16*)(ws + alloc((size_t)Cn * Cn * 2));
  bf16* w1p  = (bf16*)(ws + alloc((size_t)3 * HCHn * Cn * 2));
  bf16* w2p  = (bf16*)(ws + alloc((size_t)3 * Cn * HCHn * 2));
  bf16* qt   = (bf16*)(ws + alloc((size_t)Bn * Hn * Tn * CKn * 2));
  bf16* kt   = (bf16*)(ws + alloc((size_t)Bn * Hn * Tn * CKn * 2));
  bf16* vt   = (bf16*)(ws + alloc((size_t)Bn * Hn * CKn * Tn * 2));
  bf16* att  = (bf16*)(ws + alloc((size_t)Bn * Tn * Cn * 2));
  float* s1  = (float*)(ws + alloc((size_t)Bn * Tn * Cn * 4));
  float* h32 = (float*)(ws + alloc((size_t)Bn * Tn * Cn * 4));
  bf16* hb   = (bf16*)(ws + alloc((size_t)Bn * Tn * Cn * 2));
  bf16* fbuf = (bf16*)(ws + alloc((size_t)Bn * Tn * HCHn * 2));
  float* s2  = (float*)(ws + alloc((size_t)Bn * Tn * Cn * 4));

  // ---- prep: bf16 conversions / weight repack ----
  {
    int n = Bn * Cn * Tn;
    k_f2bf<<<(n + 255) / 256, 256, 0, stream>>>(x, xb, n);
    n = Cn * CKn;
    k_f2bf<<<(n + 255) / 256, 256, 0, stream>>>(wq, wqb, n);
    k_f2bf<<<(n + 255) / 256, 256, 0, stream>>>(wk, wkb, n);
    k_f2bf<<<(n + 255) / 256, 256, 0, stream>>>(wv, wvb, n);
    n = Cn * Cn;
    k_f2bf<<<(n + 255) / 256, 256, 0, stream>>>(wo, wob, n);
    n = 3 * HCHn * Cn;
    k_pack3<<<(n + 255) / 256, 256, 0, stream>>>(w1, w1p, HCHn, Cn);
    n = 3 * Cn * HCHn;
    k_pack3<<<(n + 255) / 256, 256, 0, stream>>>(w2, w2p, Cn, HCHn);
  }

  // ---- main pipeline ----
  k_qkv   <<<Bn * Hn * (Tn / 64), 256, 0, stream>>>(xb, wqb, wkb, wvb, bq, bk, bv, qt, kt, vt);
  k_attn  <<<Bn * Hn * (Tn / 64), 256, 0, stream>>>(qt, kt, vt, amask, att);
  k_wogemm<<<Bn * (Cn / 64) * (Tn / 64), 256, 0, stream>>>(wob, att, bo, x, s1);
  k_ln1   <<<Bn * Tn, 256, 0, stream>>>(s1, g1, be1, xmask, hb, h32);
  k_conv1 <<<Bn * (HCHn / 64) * (Tn / 64), 256, 0, stream>>>(w1p, hb, b1, xmask, fbuf);
  k_conv2 <<<Bn * (Cn / 64) * (Tn / 64), 256, 0, stream>>>(w2p, fbuf, b2, xmask, h32, s2);
  k_ln2   <<<Bn * Tn, 256, 0, stream>>>(s2, g2, be2, out);
}
